// DGCNN__47296179863981
// MI455X (gfx1250) — matrix operand order repaired and checked by reading, not compiled, not run on hardware
//
#include <hip/hip_runtime.h>
#include <hip/hip_bf16.h>
#include <math.h>

// ---------------------------------------------------------------------------
// DGCNN forward for MI455X (gfx1250, wave32).
// Heavy GEMMs (edge convs, point conv, KNN Gram) use v_wmma_f32_16x16x32_bf16.
// BN stats are two-stage deterministic (no float atomics).
// Workspace budget ~236 MB (XC 31.5M, dist 33.5M, h 168M bf16, misc).
// ---------------------------------------------------------------------------

typedef __attribute__((ext_vector_type(16))) __bf16 v16bf;
typedef __attribute__((ext_vector_type(8)))  __bf16 v8bf;
typedef __attribute__((ext_vector_type(8)))  float  v8f;

#define B_    8
#define NP    1024
#define KNN   20
#define CTOT  960
#define BNEPS 1e-5f

__device__ __forceinline__ float gelu_erf(float x) {
    return 0.5f * x * (1.0f + erff(x * 0.70710678118654752f));
}

// --------------------------- small f32 kernels -----------------------------

__global__ void sqnorm_kernel(const float* __restrict__ X, int Cin, int bstride,
                              float* __restrict__ sq) {
    int t = blockIdx.x * blockDim.x + threadIdx.x;          // b*NP + n
    if (t >= B_ * NP) return;
    int b = t / NP, n = t % NP;
    const float* xp = X + (size_t)b * bstride + n;
    float s = 0.f;
    for (int c = 0; c < Cin; ++c) { float v = xp[(size_t)c * NP]; s += v * v; }
    sq[t] = s;
}

// inner products for C=3 (block1), stores inner[b][n][m]
__global__ void dist3_kernel(const float* __restrict__ X, float* __restrict__ dist) {
    size_t t = (size_t)blockIdx.x * blockDim.x + threadIdx.x;
    if (t >= (size_t)B_ * NP * NP) return;
    int m = (int)(t % NP);
    size_t r = t / NP;
    int n = (int)(r % NP);
    int b = (int)(r / NP);
    const float* xb = X + (size_t)b * (3 * NP);
    dist[t] = xb[n] * xb[m] + xb[NP + n] * xb[NP + m] + xb[2 * NP + n] * xb[2 * NP + m];
}

// top-k by key 2*inner[n][m] - sq[m]  (equivalent ordering to reference negd)
__global__ void topk_kernel(const float* __restrict__ dist, const float* __restrict__ sq,
                            int* __restrict__ idx) {
    int t = blockIdx.x * blockDim.x + threadIdx.x;          // b*NP + n
    if (t >= B_ * NP) return;
    int b = t / NP;
    const float* row = dist + (size_t)t * NP;
    const float* sqb = sq + b * NP;
    float vals[KNN]; int inds[KNN];
#pragma unroll
    for (int i = 0; i < KNN; ++i) { vals[i] = -__builtin_inff(); inds[i] = 0; }
    for (int m = 0; m < NP; ++m) {
        float v = 2.f * row[m] - sqb[m];
        if (v > vals[0]) {
            vals[0] = v; inds[0] = m;
#pragma unroll
            for (int i = 0; i < KNN - 1; ++i) {
                if (vals[i] > vals[i + 1]) {
                    float tv = vals[i]; vals[i] = vals[i + 1]; vals[i + 1] = tv;
                    int ti = inds[i]; inds[i] = inds[i + 1]; inds[i + 1] = ti;
                }
            }
        }
    }
    int* op = idx + (size_t)t * KNN;
#pragma unroll
    for (int i = 0; i < KNN; ++i) op[i] = inds[i];
}

__global__ void f32_to_bf16_kernel(const float* __restrict__ in, __bf16* __restrict__ out, int n) {
    int t = blockIdx.x * blockDim.x + threadIdx.x;
    if (t < n) out[t] = (__bf16)in[t];
}

// block1 edge conv (K=6): trivial VALU GEMM, h stored pre-BN as bf16
__global__ void edge1_kernel(const float* __restrict__ X, const int* __restrict__ idx,
                             const float* __restrict__ W1, __bf16* __restrict__ h) {
    int t = blockIdx.x * blockDim.x + threadIdx.x;          // (b*NP + p)*KNN + j
    if (t >= B_ * NP * KNN) return;
    int j = t % KNN;
    int p = (t / KNN) % NP;
    int b = t / (NP * KNN);
    int q = idx[t];
    const float* xb = X + (size_t)b * (3 * NP);
    float f[6];
#pragma unroll
    for (int c = 0; c < 3; ++c) {
        float ctr = xb[c * NP + p];
        f[c]     = xb[c * NP + q] - ctr;
        f[3 + c] = ctr;
    }
    int cg = p * KNN + j;
    for (int o = 0; o < 64; ++o) {
        const float* w = W1 + o * 6;
        float s = w[0]*f[0] + w[1]*f[1] + w[2]*f[2] + w[3]*f[3] + w[4]*f[4] + w[5]*f[5];
        h[((size_t)(b * 64 + o)) * (NP * KNN) + cg] = (__bf16)s;
    }
}

// --------------------- WMMA Gram GEMM (KNN inner products) -----------------
// inner[b][n][m] = sum_c x[b][c][n]*x[b][c][m], bf16 inputs, f32 accumulate.
#define GP (256 + 8)   // padded channel stride in LDS (max Cin = 256)

__global__ void __launch_bounds__(128)
gram_kernel(const float* __restrict__ X, int Cin, int bstride, float* __restrict__ dist) {
    __shared__ __bf16 xs[32 * GP];        // 32 column points, channel-contiguous
    __shared__ __bf16 as[4 * 16 * GP];    // per-wave 16 row points
    int b  = blockIdx.y;
    int m0 = blockIdx.x * 32;
    const float* xb = X + (size_t)b * bstride;
    int tid = threadIdx.x;
    for (int e = tid; e < 32 * Cin; e += 128) {
        int mm = e & 31, c = e >> 5;
        xs[mm * GP + c] = (__bf16)xb[(size_t)c * NP + m0 + mm];    // coalesced over mm
    }
    __syncthreads();
    int lane = tid & 31, wid = tid >> 5;
    int g = lane >> 4, l16 = lane & 15;
    int colsub = wid & 1, mhalf = wid >> 1;
    int mcol = m0 + colsub * 16 + l16;
    __bf16* aw = as + wid * (16 * GP);
    int ksteps = Cin >> 5;
    float* dbase = dist + (size_t)b * NP * NP;
    for (int mt = mhalf; mt < NP / 16; mt += 2) {
        int n0 = mt * 16;
        for (int c = g; c < Cin; c += 2)                   // wave stages its own A tile
            aw[l16 * GP + c] = (__bf16)xb[(size_t)c * NP + n0 + l16];
        __syncthreads();                                   // uniform trip counts
        v8f acc = {};
        for (int ks = 0; ks < ksteps; ++ks) {
            int kb = ks * 32;
            const __bf16* ap = aw + l16 * GP + kb;
            v8bf alo = *(const v8bf*)(ap + 8 * g);
            v8bf ahi = *(const v8bf*)(ap + 16 + 8 * g);
            const __bf16* bp = xs + (colsub * 16 + l16) * GP + kb;
            v8bf blo = *(const v8bf*)(bp + 8 * g);
            v8bf bhi = *(const v8bf*)(bp + 16 + 8 * g);
            v16bf a, bb;
#pragma unroll
            for (int i = 0; i < 8; ++i) { a[i]=alo[i]; a[i+8]=ahi[i]; bb[i]=blo[i]; bb[i+8]=bhi[i]; }
            acc = __builtin_amdgcn_wmma_f32_16x16x32_bf16(false, a, false, bb,
                                                          (short)0, acc, false, false);
        }
#pragma unroll
        for (int r = 0; r < 8; ++r) {
            int n = n0 + r + 8 * g;
            dbase[(size_t)n * NP + mcol] = acc[r];
        }
    }
}

// ------------- WMMA gather-GEMM: h = W @ edge_feature (pre-BN) -------------
// kneigh>1: columns are (p,j), f = concat(X[:,q]-X[:,p], X[:,p]), K2=2*Cin.
// kneigh==1: plain point conv, f = X[:,p], K2=Cin.
// LDS tile: 32 cols x K2 halves, col stride 1024 halves, 16-half chunks
// XOR-swizzled by (col&7) to spread banks at the 2048B stride.
__global__ void __launch_bounds__(128)
wmma_gemm_kernel(const float* __restrict__ X, int bstride, int Cin, int K2,
                 const int* __restrict__ idx, int kneigh,
                 const __bf16* __restrict__ Wb, int O,
                 __bf16* __restrict__ h) {
    __shared__ __bf16 fs[32 * 1024];                       // 64 KB
    int Nk   = NP * kneigh;
    int col0 = blockIdx.x * 32;
    int b    = col0 / Nk;                                  // Nk is a multiple of 32
    int rb   = col0 % Nk;
    const float* xb = X + (size_t)b * bstride;
    int tid = threadIdx.x;
    {                                                      // gather phase: 4 threads/column
        int c     = tid & 31;
        int klane = tid >> 5;
        int cg = rb + c;
        int p  = cg / kneigh;
        int j  = cg - p * kneigh;
        int q  = (kneigh > 1) ? idx[((size_t)b * NP + p) * kneigh + j] : p;
        for (int kk = klane; kk < K2; kk += 4) {
            float v;
            if (kneigh > 1) {
                if (kk < Cin) v = xb[(size_t)kk * NP + q] - xb[(size_t)kk * NP + p];
                else          v = xb[(size_t)(kk - Cin) * NP + p];
            } else {
                v = xb[(size_t)kk * NP + p];
            }
            int chunk = (kk >> 4) ^ (c & 7);
            fs[c * 1024 + chunk * 16 + (kk & 15)] = (__bf16)v;
        }
    }
    __syncthreads();
    int lane = tid & 31, wid = tid >> 5;
    int g = lane >> 4, l16 = lane & 15;
    int colsub = wid & 1, mhalf = wid >> 1;
    int c_local = colsub * 16 + l16;
    int cg_out  = rb + c_local;
    int ksteps  = K2 >> 5;
    int MT      = O >> 5;                                  // M-tiles per wave
    const __bf16* fscol = fs + c_local * 1024;
    for (int t = 0; t < MT; ++t) {
        int mt = mhalf + 2 * t;
        const __bf16* wrow = Wb + (size_t)(mt * 16 + l16) * K2;
        v8f acc = {};
        for (int ks = 0; ks < ksteps; ++ks) {
            int kb = ks * 32;
            v8bf alo = *(const v8bf*)(wrow + kb + 16 * g);
            v8bf ahi = *(const v8bf*)(wrow + kb + 16 * g + 8);
            int chunk = ((kb + 16 * g) >> 4) ^ (c_local & 7);
            const __bf16* bp = fscol + chunk * 16;
            v8bf blo = *(const v8bf*)(bp);
            v8bf bhi = *(const v8bf*)(bp + 8);
            v16bf a, bb;
#pragma unroll
            for (int i = 0; i < 8; ++i) { a[i]=alo[i]; a[i+8]=ahi[i]; bb[i]=blo[i]; bb[i+8]=bhi[i]; }
            acc = __builtin_amdgcn_wmma_f32_16x16x32_bf16(false, a, false, bb,
                                                          (short)0, acc, false, false);
        }
#pragma unroll
        for (int r = 0; r < 8; ++r) {
            int o = mt * 16 + r + 8 * g;
            h[((size_t)b * O + o) * Nk + cg_out] = (__bf16)acc[r];
        }
    }
}

// -------------------- deterministic BN stats + epilogues -------------------

// per-(channel,batch) partial sums: part[o*B_+b], part[O*B_ + o*B_+b]
__global__ void stats_kernel(const __bf16* __restrict__ h, int O, int Nk,
                             float* __restrict__ part) {
    __shared__ float ls[256], lss[256];
    int o = blockIdx.x, b = blockIdx.y;
    const __bf16* hp = h + ((size_t)b * O + o) * Nk;
    float s = 0.f, ss = 0.f;
    for (int i = threadIdx.x; i < Nk; i += 256) {
        float v = (float)hp[i]; s += v; ss += v * v;
    }
    ls[threadIdx.x] = s; lss[threadIdx.x] = ss;
    __syncthreads();
    for (int off = 128; off > 0; off >>= 1) {
        if (threadIdx.x < off) {
            ls[threadIdx.x]  += ls[threadIdx.x + off];
            lss[threadIdx.x] += lss[threadIdx.x + off];
        }
        __syncthreads();
    }
    if (threadIdx.x == 0) {
        part[(size_t)o * B_ + b]                  = ls[0];
        part[(size_t)O * B_ + (size_t)o * B_ + b] = lss[0];
    }
}

__global__ void bn_finalize_kernel(const float* __restrict__ part, int O, float cnt,
                                   const float* __restrict__ gamma,
                                   const float* __restrict__ beta,
                                   float* __restrict__ ss) {     // ss: scale[O],shift[O]
    int o = blockIdx.x * blockDim.x + threadIdx.x;
    if (o >= O) return;
    float s = 0.f, q = 0.f;
    for (int b = 0; b < B_; ++b) {
        s += part[(size_t)o * B_ + b];
        q += part[(size_t)O * B_ + (size_t)o * B_ + b];
    }
    float mean = s / cnt;
    float var  = q / cnt - mean * mean;
    float sc   = gamma[o] * rsqrtf(var + BNEPS);
    ss[o]     = sc;
    ss[O + o] = beta[o] - mean * sc;
}

// BN + GELU + max over k neighbors -> channel slice of XC (out pre-offset)
__global__ void apply_max_kernel(const __bf16* __restrict__ h, const float* __restrict__ ss,
                                 int O, float* __restrict__ out) {
    int t = blockIdx.x * blockDim.x + threadIdx.x;        // (b*O + o)*NP + p
    if (t >= B_ * O * NP) return;
    int p = t % NP;
    int o = (t / NP) % O;
    int b = t / (NP * O);
    const __bf16* hp = h + ((size_t)b * O + o) * ((size_t)NP * KNN) + (size_t)p * KNN;
    float sc = ss[o], sh = ss[O + o];
    float m = -__builtin_inff();
#pragma unroll
    for (int j = 0; j < KNN; ++j)
        m = fmaxf(m, gelu_erf((float)hp[j] * sc + sh));
    out[(size_t)b * CTOT * NP + (size_t)o * NP + p] = m;
}

// block5: BN + GELU + (max, mean) pool over N -> z (B, 2048)
__global__ void pool5_kernel(const __bf16* __restrict__ h, const float* __restrict__ ss,
                             float* __restrict__ z) {
    __shared__ float lmax[256], lsum[256];
    int o = blockIdx.x, b = blockIdx.y;                   // O = 1024
    const __bf16* hp = h + ((size_t)b * 1024 + o) * NP;
    float sc = ss[o], sh = ss[1024 + o];
    float m = -__builtin_inff(), s = 0.f;
    for (int i = threadIdx.x; i < NP; i += 256) {
        float v = gelu_erf((float)hp[i] * sc + sh);
        m = fmaxf(m, v); s += v;
    }
    lmax[threadIdx.x] = m; lsum[threadIdx.x] = s;
    __syncthreads();
    for (int off = 128; off > 0; off >>= 1) {
        if (threadIdx.x < off) {
            lmax[threadIdx.x] = fmaxf(lmax[threadIdx.x], lmax[threadIdx.x + off]);
            lsum[threadIdx.x] += lsum[threadIdx.x + off];
        }
        __syncthreads();
    }
    if (threadIdx.x == 0) {
        z[(size_t)b * 2048 + o]        = lmax[0];
        z[(size_t)b * 2048 + 1024 + o] = lsum[0] * (1.f / NP);
    }
}

// --------------------------------- head ------------------------------------

__global__ void linear_kernel(const float* __restrict__ in, const float* __restrict__ W,
                              const float* __restrict__ bias, float* __restrict__ out,
                              int I, int O) {
    int t = blockIdx.x * blockDim.x + threadIdx.x;        // b*O + o
    if (t >= B_ * O) return;
    int o = t % O, b = t / O;
    const float* ip = in + (size_t)b * I;
    const float* wp = W + (size_t)o * I;
    float s = bias ? bias[o] : 0.f;
    for (int i = 0; i < I; ++i) s += ip[i] * wp[i];
    out[t] = s;
}

__global__ void bnvec_kernel(const float* __restrict__ z, int O,
                             const float* __restrict__ gamma, const float* __restrict__ beta,
                             float* __restrict__ ss) {
    int o = blockIdx.x * blockDim.x + threadIdx.x;
    if (o >= O) return;
    float s = 0.f, q = 0.f;
    for (int b = 0; b < B_; ++b) { float v = z[(size_t)b * O + o]; s += v; q += v * v; }
    float mean = s / B_;
    float var  = q / B_ - mean * mean;
    float sc   = gamma[o] * rsqrtf(var + BNEPS);
    ss[o] = sc; ss[O + o] = beta[o] - mean * sc;
}

__global__ void leaky_apply_kernel(float* z, const float* __restrict__ ss, int O) {
    int t = blockIdx.x * blockDim.x + threadIdx.x;
    if (t >= B_ * O) return;
    int o = t % O;
    float v = z[t] * ss[o] + ss[O + o];
    z[t] = v > 0.f ? v : 0.2f * v;
}

// ------------------------------- launcher ----------------------------------

extern "C" void kernel_launch(void* const* d_in, const int* in_sizes, int n_in,
                              void* d_out, int out_size, void* d_ws, size_t ws_size,
                              hipStream_t stream) {
    (void)in_sizes; (void)n_in; (void)out_size; (void)ws_size;

    const float* x   = (const float*)d_in[0];
    const float* W1  = (const float*)d_in[2];
    const float* g1  = (const float*)d_in[3];
    const float* b1  = (const float*)d_in[4];
    const float* W5  = (const float*)d_in[14];
    const float* g5  = (const float*)d_in[15];
    const float* b5  = (const float*)d_in[16];
    const float* Wl1 = (const float*)d_in[17];
    const float* g6  = (const float*)d_in[18];
    const float* b6  = (const float*)d_in[19];
    const float* Wl2 = (const float*)d_in[20];
    const float* bl2 = (const float*)d_in[21];
    const float* g7  = (const float*)d_in[22];
    const float* b7  = (const float*)d_in[23];
    const float* Wl3 = (const float*)d_in[24];
    const float* bl3 = (const float*)d_in[25];

    // workspace carve
    char* wsp = (char*)d_ws;
    size_t off = 0;
    auto carve = [&](size_t bytes) -> void* {
        void* p = wsp + off;
        off = (off + bytes + 255) & ~(size_t)255;
        return p;
    };
    float*  XC   = (float*)carve((size_t)B_ * CTOT * NP * 4);          // 31.5 MB
    float*  dist = (float*)carve((size_t)B_ * NP * NP * 4);            // 33.6 MB
    float*  sq   = (float*)carve((size_t)B_ * NP * 4);
    int*    idx  = (int*)  carve((size_t)B_ * NP * KNN * 4);
    __bf16* h    = (__bf16*)carve((size_t)B_ * 512 * NP * KNN * 2);    // 168 MB
    __bf16* Wb   = (__bf16*)carve((size_t)1024 * 960 * 2);
    float*  part = (float*)carve((size_t)2 * 1024 * B_ * 4);
    float*  ss   = (float*)carve((size_t)2 * 1024 * 4);
    float*  z    = (float*)carve((size_t)B_ * 2048 * 4);
    float*  z1   = (float*)carve((size_t)B_ * 512 * 4);
    float*  z2   = (float*)carve((size_t)B_ * 256 * 4);

    const int Nk = NP * KNN;

    // ---------------- block 1 (C=3, O=64) : VALU path ----------------
    sqnorm_kernel<<<(B_*NP + 255)/256, 256, 0, stream>>>(x, 3, 3*NP, sq);
    dist3_kernel<<<(int)(((size_t)B_*NP*NP + 255)/256), 256, 0, stream>>>(x, dist);
    topk_kernel<<<(B_*NP + 127)/128, 128, 0, stream>>>(dist, sq, idx);
    edge1_kernel<<<(B_*NP*KNN + 255)/256, 256, 0, stream>>>(x, idx, W1, h);
    stats_kernel<<<dim3(64, B_), 256, 0, stream>>>(h, 64, Nk, part);
    bn_finalize_kernel<<<1, 64, 0, stream>>>(part, 64, (float)((size_t)B_*Nk), g1, b1, ss);
    apply_max_kernel<<<(B_*64*NP + 255)/256, 256, 0, stream>>>(h, ss, 64, XC + 0*NP);

    // ---------------- blocks 2..4 : WMMA path ----------------
    const int cins[3]  = {64, 128, 256};
    const int os[3]    = {128, 256, 512};
    const int offi[3]  = {0, 64, 192};
    const int offo[3]  = {64, 192, 448};
    for (int blk = 0; blk < 3; ++blk) {
        int Cin = cins[blk], O = os[blk];
        const float* Wf = (const float*)d_in[5 + 3*blk];
        const float* gb = (const float*)d_in[6 + 3*blk];
        const float* bb = (const float*)d_in[7 + 3*blk];
        const float* Xin = XC + (size_t)offi[blk] * NP;

        sqnorm_kernel<<<(B_*NP + 255)/256, 256, 0, stream>>>(Xin, Cin, CTOT*NP, sq);
        gram_kernel<<<dim3(NP/32, B_), 128, 0, stream>>>(Xin, Cin, CTOT*NP, dist);
        topk_kernel<<<(B_*NP + 127)/128, 128, 0, stream>>>(dist, sq, idx);
        f32_to_bf16_kernel<<<(O*2*Cin + 255)/256, 256, 0, stream>>>(Wf, Wb, O*2*Cin);
        wmma_gemm_kernel<<<(B_*Nk)/32, 128, 0, stream>>>(Xin, CTOT*NP, Cin, 2*Cin,
                                                         idx, KNN, Wb, O, h);
        stats_kernel<<<dim3(O, B_), 256, 0, stream>>>(h, O, Nk, part);
        bn_finalize_kernel<<<(O + 63)/64, 64, 0, stream>>>(part, O, (float)((size_t)B_*Nk),
                                                           gb, bb, ss);
        apply_max_kernel<<<(B_*O*NP + 255)/256, 256, 0, stream>>>(h, ss, O,
                                                                  XC + (size_t)offo[blk]*NP);
    }

    // ---------------- block 5 : 1024x960 point conv (WMMA) ----------------
    f32_to_bf16_kernel<<<(1024*960 + 255)/256, 256, 0, stream>>>(W5, Wb, 1024*960);
    wmma_gemm_kernel<<<(B_*NP)/32, 128, 0, stream>>>(XC, CTOT*NP, 960, 960,
                                                     nullptr, 1, Wb, 1024, h);
    stats_kernel<<<dim3(1024, B_), 256, 0, stream>>>(h, 1024, NP, part);
    bn_finalize_kernel<<<16, 64, 0, stream>>>(part, 1024, (float)(B_*NP), g5, b5, ss);
    pool5_kernel<<<dim3(1024, B_), 256, 0, stream>>>(h, ss, z);

    // ---------------- head MLP ----------------
    linear_kernel<<<(B_*512 + 255)/256, 256, 0, stream>>>(z, Wl1, nullptr, z1, 2048, 512);
    bnvec_kernel<<<2, 256, 0, stream>>>(z1, 512, g6, b6, ss);
    leaky_apply_kernel<<<(B_*512 + 255)/256, 256, 0, stream>>>(z1, ss, 512);
    linear_kernel<<<(B_*256 + 255)/256, 256, 0, stream>>>(z1, Wl2, bl2, z2, 512, 256);
    bnvec_kernel<<<1, 256, 0, stream>>>(z2, 256, g7, b7, ss);
    leaky_apply_kernel<<<(B_*256 + 255)/256, 256, 0, stream>>>(z2, ss, 256);
    linear_kernel<<<(B_*40 + 255)/256, 256, 0, stream>>>(z2, Wl3, bl3, (float*)d_out, 256, 40);
}
